// DH_DendriticLayer_2637109920471
// MI455X (gfx1250) — matrix operand order repaired and checked by reading, not compiled, not run on hardware
//
#include <hip/hip_runtime.h>
#include <hip/hip_bf16.h>

// Dendritic-branch SNN layer fused kernel for gfx1250 (MI455X).
// Block-masked matmul == 4 dense GEMMs (M=1024, N=2048, K=1024), computed with
// V_WMMA_F32_16X16X32_F16 (f16 inputs, f32 accumulation). A (spikes) is exactly
// 0/1 -> f16-lossless; per-output branch scale (1-beta[o,j]) is folded into the
// W tile during the f32->f16 LDS staging pass, so one accumulator sums all 4
// branch GEMMs. LIF epilogue fused.

typedef __attribute__((ext_vector_type(16))) _Float16 v16h;
typedef __attribute__((ext_vector_type(8)))  _Float16 v8h;
typedef __attribute__((ext_vector_type(8)))  float    v8f;

#define B_DIM   1024
#define IN_DIM  4096
#define OUT_DIM 2048
#define NBR     4
#define S_DIM   1024            // IN_DIM / NBR
#define BM      128
#define BN      128
#define KT      64              // K-chunk per LDS stage (two wmma k-steps)
#define LDH     72              // padded f16 row stride: 144B, 16B aligned, conflict-free
#define NCHUNK  (NBR * (S_DIM / KT))   // 64 stages total

__device__ __forceinline__ float sigmoidf_dev(float x) {
    return 1.0f / (1.0f + __expf(-x));
}

__device__ __forceinline__ v16h cat8(v8h lo, v8h hi) {
    return __builtin_shufflevector(lo, hi, 0,1,2,3,4,5,6,7,8,9,10,11,12,13,14,15);
}

__global__ __launch_bounds__(256)
void DH_DendriticLayer_fused(const float* __restrict__ input_spike,  // [B, IN]
                             const float* __restrict__ mem,          // [B, OUT]
                             const float* __restrict__ spike,        // [B, OUT]
                             const float* __restrict__ d_input,      // [B, OUT, 4]
                             const float* __restrict__ W,            // [OUT*4, IN]
                             const float* __restrict__ bias,         // [OUT*4]
                             const float* __restrict__ tau_m,        // [OUT]
                             const float* __restrict__ tau_n,        // [OUT, 4]
                             float* __restrict__ out)                // [2, B, OUT]
{
    // double-buffered f16 tiles: [buf][row][K]
    __shared__ __align__(16) _Float16 As[2][BM * LDH];
    __shared__ __align__(16) _Float16 Bs[2][BN * LDH];
    __shared__ __align__(16) float    Ep[6 * 128];   // alpha,biasP,beta[4] per column

    const int tid    = threadIdx.x;
    const int lane   = tid & 31;
    const int wave   = tid >> 5;
    const int wave_m = wave & 3;       // 4 waves along M  (4*32 = 128)
    const int wave_n = wave >> 2;      // 2 waves along N  (2*64 = 128)
    const int laneLo = lane & 15;
    const int laneHi = lane >> 4;

    const int m0 = blockIdx.y * BM;
    const int n0 = blockIdx.x * BN;

    // staging: 2 threads per row, 32 floats (=32 halves) each
    const int rowS = tid >> 1;         // 0..127
    const int half = tid & 1;
    const int sbase = rowS * LDH + half * 32;

    // per-thread (1-beta) scales for the W row this thread stages, one per branch
    float sBv[NBR];
    #pragma unroll
    for (int j = 0; j < NBR; ++j)
        sBv[j] = 1.0f - sigmoidf_dev(tau_n[(n0 + rowS) * NBR + j]);

    v8f acc[2][4];
    #pragma unroll
    for (int i = 0; i < 2; ++i)
        #pragma unroll
        for (int t = 0; t < 4; ++t)
            acc[i][t] = (v8f){0.f,0.f,0.f,0.f,0.f,0.f,0.f,0.f};

    // chunk c -> branch j = c/16, k-offset kc = (c%16)*KT
    auto a_ptr = [&](int c) {
        const int j = c >> 4, kc = (c & 15) * KT;
        return input_spike + (size_t)(m0 + rowS) * IN_DIM + j * S_DIM + kc + half * 32;
    };
    auto w_ptr = [&](int c) {
        const int j = c >> 4, kc = (c & 15) * KT;
        return W + ((size_t)(n0 + rowS) * NBR + j) * IN_DIM + j * S_DIM + kc + half * 32;
    };

    float4 aR[8], wR[8];

    // ---- prologue: stage chunk 0 into buffer 0 ----
    {
        const float* Ap = a_ptr(0);
        const float* Wp = w_ptr(0);
        #pragma unroll
        for (int q = 0; q < 8; ++q) {
            aR[q] = *(const float4*)(Ap + q * 4);
            wR[q] = *(const float4*)(Wp + q * 4);
        }
        const float sB = sBv[0];
        #pragma unroll
        for (int q = 0; q < 4; ++q) {
            const float4 a0 = aR[2*q], a1 = aR[2*q+1];
            const float4 w0 = wR[2*q], w1 = wR[2*q+1];
            v8h av, wv;
            av[0]=(_Float16)a0.x; av[1]=(_Float16)a0.y; av[2]=(_Float16)a0.z; av[3]=(_Float16)a0.w;
            av[4]=(_Float16)a1.x; av[5]=(_Float16)a1.y; av[6]=(_Float16)a1.z; av[7]=(_Float16)a1.w;
            wv[0]=(_Float16)(w0.x*sB); wv[1]=(_Float16)(w0.y*sB); wv[2]=(_Float16)(w0.z*sB); wv[3]=(_Float16)(w0.w*sB);
            wv[4]=(_Float16)(w1.x*sB); wv[5]=(_Float16)(w1.y*sB); wv[6]=(_Float16)(w1.z*sB); wv[7]=(_Float16)(w1.w*sB);
            *(v8h*)&As[0][sbase + q * 8] = av;
            *(v8h*)&Bs[0][sbase + q * 8] = wv;
        }
    }
    __syncthreads();

    // ---- main pipeline: one barrier per stage, LDS double-buffered ----
    for (int c = 0; c < NCHUNK; ++c) {
        const int p = c & 1;
        const bool haveNext = (c + 1) < NCHUNK;

        // issue global loads for chunk c+1 (hidden under the wmma burst below)
        if (haveNext) {
            const float* Ap = a_ptr(c + 1);
            const float* Wp = w_ptr(c + 1);
            #pragma unroll
            for (int q = 0; q < 8; ++q) {
                aR[q] = *(const float4*)(Ap + q * 4);
                wR[q] = *(const float4*)(Wp + q * 4);
            }
            if (c + 2 < NCHUNK) {                // deep prefetch into L2/WGP$
                __builtin_prefetch(a_ptr(c + 2), 0, 0);
                __builtin_prefetch(w_ptr(c + 2), 0, 0);
            }
        }

        // compute chunk c from buffer p
        #pragma unroll
        for (int kk = 0; kk < KT; kk += 32) {
            v16h aF[2], bF[4];
            #pragma unroll
            for (int i = 0; i < 2; ++i) {
                const _Float16* pa =
                    &As[p][(wave_m * 32 + i * 16 + laneLo) * LDH + kk + laneHi * 8];
                aF[i] = cat8(*(const v8h*)pa, *(const v8h*)(pa + 16));
            }
            #pragma unroll
            for (int t = 0; t < 4; ++t) {
                const _Float16* pb =
                    &Bs[p][(wave_n * 64 + t * 16 + laneLo) * LDH + kk + laneHi * 8];
                bF[t] = cat8(*(const v8h*)pb, *(const v8h*)(pb + 16));
            }
            #pragma unroll
            for (int i = 0; i < 2; ++i)
                #pragma unroll
                for (int t = 0; t < 4; ++t)
                    acc[i][t] = __builtin_amdgcn_wmma_f32_16x16x32_f16(
                        /*neg_a=*/false, aF[i], /*neg_b=*/false, bF[t],
                        /*c_mod=*/(short)0, acc[i][t],
                        /*reuse_a=*/false, /*reuse_b=*/false);
        }

        // stage chunk c+1 into buffer 1-p (safe: prior reads of 1-p fenced by
        // the barrier at the end of iteration c-1)
        if (haveNext) {
            const float sB = sBv[(c + 1) >> 4];
            #pragma unroll
            for (int q = 0; q < 4; ++q) {
                const float4 a0 = aR[2*q], a1 = aR[2*q+1];
                const float4 w0 = wR[2*q], w1 = wR[2*q+1];
                v8h av, wv;
                av[0]=(_Float16)a0.x; av[1]=(_Float16)a0.y; av[2]=(_Float16)a0.z; av[3]=(_Float16)a0.w;
                av[4]=(_Float16)a1.x; av[5]=(_Float16)a1.y; av[6]=(_Float16)a1.z; av[7]=(_Float16)a1.w;
                wv[0]=(_Float16)(w0.x*sB); wv[1]=(_Float16)(w0.y*sB); wv[2]=(_Float16)(w0.z*sB); wv[3]=(_Float16)(w0.w*sB);
                wv[4]=(_Float16)(w1.x*sB); wv[5]=(_Float16)(w1.y*sB); wv[6]=(_Float16)(w1.z*sB); wv[7]=(_Float16)(w1.w*sB);
                *(v8h*)&As[1 - p][sbase + q * 8] = av;
                *(v8h*)&Bs[1 - p][sbase + q * 8] = wv;
            }
        }
        __syncthreads();
    }

    // ---- epilogue: per-output-column LIF params into LDS ----
    float* ep_alpha = Ep;            // [128]
    float* ep_biasP = Ep + 128;      // [128]  sum_j (1-beta_j)*b_j
    float* ep_beta  = Ep + 256;      // [4][128]
    if (tid < BN) {
        const int o = n0 + tid;
        ep_alpha[tid] = sigmoidf_dev(tau_m[o]);
        float bp = 0.0f;
        #pragma unroll
        for (int j = 0; j < NBR; ++j) {
            const float be = sigmoidf_dev(tau_n[o * NBR + j]);
            ep_beta[j * 128 + tid] = be;
            bp += (1.0f - be) * bias[o * NBR + j];
        }
        ep_biasP[tid] = bp;
    }
    __syncthreads();

    float* out_mem   = out;
    float* out_spike = out + (size_t)B_DIM * OUT_DIM;

    #pragma unroll
    for (int i = 0; i < 2; ++i) {
        #pragma unroll
        for (int t = 0; t < 4; ++t) {
            const int col = wave_n * 64 + t * 16 + laneLo;   // 0..127 in tile
            const int o   = n0 + col;
            const float al  = ep_alpha[col];
            const float omA = 1.0f - al;
            const float bp  = ep_biasP[col];
            const float be0 = ep_beta[0 * 128 + col];
            const float be1 = ep_beta[1 * 128 + col];
            const float be2 = ep_beta[2 * 128 + col];
            const float be3 = ep_beta[3 * 128 + col];
            const int mbase = m0 + wave_m * 32 + i * 16 + laneHi * 8;
            #pragma unroll
            for (int v = 0; v < 8; ++v) {
                const size_t idx = (size_t)(mbase + v) * OUT_DIM + o;
                const float4 d4 = *(const float4*)(d_input + idx * 4);
                const float dterm = be0 * d4.x + be1 * d4.y + be2 * d4.z + be3 * d4.w;
                const float l  = acc[i][t][v] + bp + dterm;
                const float mn = mem[idx] * al + omA * l - spike[idx];  // VTH=1
                out_mem[idx]   = mn;
                out_spike[idx] = (mn - 1.0f > 0.0f) ? 1.0f : 0.0f;
            }
        }
    }
}

extern "C" void kernel_launch(void* const* d_in, const int* in_sizes, int n_in,
                              void* d_out, int out_size, void* d_ws, size_t ws_size,
                              hipStream_t stream) {
    (void)in_sizes; (void)n_in; (void)out_size; (void)d_ws; (void)ws_size;
    const float* input_spike = (const float*)d_in[0];
    const float* mem         = (const float*)d_in[1];
    const float* spike       = (const float*)d_in[2];
    const float* d_input     = (const float*)d_in[3];
    const float* W           = (const float*)d_in[4];
    const float* bias        = (const float*)d_in[5];
    const float* tau_m       = (const float*)d_in[6];
    const float* tau_n       = (const float*)d_in[7];
    float* out = (float*)d_out;

    dim3 grid(OUT_DIM / BN, B_DIM / BM, 1);   // (16, 8) = 128 workgroups
    dim3 block(256, 1, 1);                    // 8 wave32s
    DH_DendriticLayer_fused<<<grid, block, 0, stream>>>(
        input_spike, mem, spike, d_input, W, bias, tau_m, tau_n, out);
}
